// SimpleFP4Model_77996606095676
// MI455X (gfx1250) — matrix operand order repaired and checked by reading, not compiled, not run on hardware
//
#include <hip/hip_runtime.h>

// ---------------------------------------------------------------------------
// Types for CDNA5 WMMA (wave32, bf16 16x16x32 -> f32 accum)
// ---------------------------------------------------------------------------
typedef __attribute__((ext_vector_type(16))) __bf16 v16bf;
typedef __attribute__((ext_vector_type(8)))  __bf16 v8bf;
typedef __attribute__((ext_vector_type(8)))  float  v8f;

#define TBM 128
#define TBN 128
#define TBK 64
#define LDS_STRIDE 72   // 64 K-elems + 8 pad; 72*2=144B rows keep b128 alignment

// ---------------------------------------------------------------------------
// CDNA5 async global->LDS copy (ASYNCcnt path, no VGPR staging).
// LDS destination address is the low 32 bits of the generic LDS pointer
// (ISA: generic->LDS aperture mapping uses addr[31:0]).
// ---------------------------------------------------------------------------
__device__ __forceinline__ void async_copy_b128(const void* gaddr, void* lds_ptr) {
  const unsigned lds_addr = (unsigned)(unsigned long long)lds_ptr;
  asm volatile("global_load_async_to_lds_b128 %0, %1, off"
               :
               : "v"(lds_addr), "v"(gaddr)
               : "memory");
}

__device__ __forceinline__ void wait_async_all() {
  asm volatile("s_wait_asynccnt 0x0" ::: "memory");
}

// ---------------------------------------------------------------------------
// Kernel 1: simulated FP4 e2m1 quant-dequant of W1, output bf16.
// One 256-thread block per output channel (row of W1).
// ---------------------------------------------------------------------------
__global__ __launch_bounds__(256) void k_dequant_fp4_w1(
    const float* __restrict__ W, __bf16* __restrict__ out, int K) {
  const int row = blockIdx.x;
  const float* w = W + (size_t)row * (size_t)K;

  float m = 0.0f;
  for (int i = threadIdx.x; i < K; i += 256) m = fmaxf(m, fabsf(w[i]));

  __shared__ float red[256];
  red[threadIdx.x] = m;
  __syncthreads();
  for (int s = 128; s > 0; s >>= 1) {
    if ((int)threadIdx.x < s)
      red[threadIdx.x] = fmaxf(red[threadIdx.x], red[threadIdx.x + s]);
    __syncthreads();
  }
  const float scale = fmaxf(red[0] * (1.0f / 6.0f), 1e-12f);
  const float inv = 1.0f / scale;

  __bf16* o = out + (size_t)row * (size_t)K;
  for (int i = threadIdx.x; i < K; i += 256) {
    const float v = w[i];
    const float a = fabsf(v) * inv;   // in [0, 6]
    float q;
    if      (a <= 0.25f) q = 0.0f;    // searchsorted 'left' boundary semantics
    else if (a <= 0.75f) q = 0.5f;
    else if (a <= 1.25f) q = 1.0f;
    else if (a <= 1.75f) q = 1.5f;
    else if (a <= 2.5f)  q = 2.0f;
    else if (a <= 3.5f)  q = 3.0f;
    else if (a <= 5.0f)  q = 4.0f;
    else                 q = 6.0f;
    o[i] = (__bf16)copysignf(q * scale, v);
  }
}

// ---------------------------------------------------------------------------
// Small converts
// ---------------------------------------------------------------------------
__global__ __launch_bounds__(256) void k_f32_to_bf16(
    const float* __restrict__ in, __bf16* __restrict__ out, int n) {
  for (int i = blockIdx.x * 256 + threadIdx.x; i < n; i += gridDim.x * 256)
    out[i] = (__bf16)in[i];
}

__global__ __launch_bounds__(256) void k_bf16_to_f32(
    const __bf16* __restrict__ in, float* __restrict__ out, int n) {
  int i = blockIdx.x * 256 + threadIdx.x;
  if (i < n) out[i] = (float)in[i];
}

// ---------------------------------------------------------------------------
// Tiled bf16 WMMA GEMM:  C[M,Ncols] = act[M,K] * weight[Ncols,K]^T + bias
// Block: 256 threads (8 waves). Block tile 128x128, K-tile 64, double-buffered
// LDS filled by async global->LDS copies overlapped with WMMA compute.
// Wave (wr=w>>2, wc=w&3) owns a 64x32 sub-tile = 4x2 16x16 WMMA accumulators.
// Assumes M%128==0, Ncols%128==0, K%128==0 (true for this problem).
// ---------------------------------------------------------------------------
template <bool RELU>
__global__ __launch_bounds__(256) void k_gemm_bf16_wmma(
    const __bf16* __restrict__ A,     // [M, K] activations, row-major
    const __bf16* __restrict__ B,     // [Ncols, K] weights, row-major
    const float*  __restrict__ bias,  // [Ncols]
    __bf16* __restrict__ C,           // [M, Ncols]
    int M, int Ncols, int K) {
  __shared__ __align__(16) __bf16 As[2][TBM * LDS_STRIDE];
  __shared__ __align__(16) __bf16 Bs[2][TBN * LDS_STRIDE];

  const int t    = threadIdx.x;
  const int wave = t >> 5;
  const int lane = t & 31;
  const int lr   = lane & 15;
  const int lh   = lane >> 4;          // 0 for lanes 0-15, 1 for 16-31

  const int rowBase = blockIdx.y * TBM;
  const int colBase = blockIdx.x * TBN;
  const int wm = (wave >> 2) * 64;     // wave row offset inside tile: 0 or 64
  const int wn = (wave & 3) * 32;      // wave col offset inside tile: 0..96

  // Issue async fills for one double-buffer slot (128x64 bf16 per matrix).
  auto issue_tile = [&](int buf, int k0) {
#pragma unroll
    for (int it = 0; it < 4; ++it) {
      const int L = t + 256 * it;      // 0..1023, 8 chunks of 8 elems per row
      const int r = L >> 3;
      const int c = (L & 7) * 8;
      async_copy_b128(A + (size_t)(rowBase + r) * (size_t)K + k0 + c,
                      &As[buf][r * LDS_STRIDE + c]);
      async_copy_b128(B + (size_t)(colBase + r) * (size_t)K + k0 + c,
                      &Bs[buf][r * LDS_STRIDE + c]);
    }
  };

  v8f acc[4][2];
#pragma unroll
  for (int i = 0; i < 4; ++i)
#pragma unroll
    for (int j = 0; j < 2; ++j)
#pragma unroll
      for (int r = 0; r < 8; ++r) acc[i][j][r] = 0.0f;

  // Prologue: fill buffer 0.
  issue_tile(0, 0);
  wait_async_all();
  __syncthreads();

  int cur = 0;
  for (int k0 = 0; k0 < K; k0 += TBK) {
    // Prefetch next K-tile into the other buffer while we compute.
    if (k0 + TBK < K) issue_tile(cur ^ 1, k0 + TBK);

#pragma unroll
    for (int ks = 0; ks < TBK; ks += 32) {
      // A fragments: 16x32 bf16 per the ISA layout.
      // lane<16: row M=lr, K = {ks+0..7, ks+16..23}
      // lane>=16: row M=lr, K = {ks+8..15, ks+24..31}
      v16bf afrag[4];
#pragma unroll
      for (int i = 0; i < 4; ++i) {
        const __bf16* p =
            &As[cur][(wm + i * 16 + lr) * LDS_STRIDE + ks + lh * 8];
        const v8bf lo = *(const v8bf*)p;
        const v8bf hi = *(const v8bf*)(p + 16);
#pragma unroll
        for (int e = 0; e < 8; ++e) { afrag[i][e] = lo[e]; afrag[i][e + 8] = hi[e]; }
      }
      // B fragments: 32x16 bf16. lane<16: col N=lr, K={ks+0..7, ks+16..23};
      // lane>=16: col N=lr, K={ks+8..15, ks+24..31}. B col n == weight row n.
      v16bf bfrag[2];
#pragma unroll
      for (int j = 0; j < 2; ++j) {
        const __bf16* p =
            &Bs[cur][(wn + j * 16 + lr) * LDS_STRIDE + ks + lh * 8];
        const v8bf lo = *(const v8bf*)p;
        const v8bf hi = *(const v8bf*)(p + 16);
#pragma unroll
        for (int e = 0; e < 8; ++e) { bfrag[j][e] = lo[e]; bfrag[j][e + 8] = hi[e]; }
      }
#pragma unroll
      for (int i = 0; i < 4; ++i)
#pragma unroll
        for (int j = 0; j < 2; ++j)
          acc[i][j] = __builtin_amdgcn_wmma_f32_16x16x32_bf16(
              /*neg_a=*/false, afrag[i], /*neg_b=*/false, bfrag[j],
              /*c_mod=*/(short)0, acc[i][j], /*reuse_a=*/false, /*reuse_b=*/false);
    }

    // Ensure our prefetch landed, then sync the workgroup before swapping.
    wait_async_all();
    __syncthreads();
    cur ^= 1;
  }

  // Epilogue: C VGPR layout -> element (r, lane): M = lh*8 + r, N = lr
#pragma unroll
  for (int i = 0; i < 4; ++i) {
#pragma unroll
    for (int j = 0; j < 2; ++j) {
      const int col = colBase + wn + j * 16 + lr;
      const float bv = bias[col];
#pragma unroll
      for (int r = 0; r < 8; ++r) {
        const int row = rowBase + wm + i * 16 + lh * 8 + r;
        float v = acc[i][j][r] + bv;
        if (RELU) v = fmaxf(v, 0.0f);
        C[(size_t)row * (size_t)Ncols + col] = (__bf16)v;
      }
    }
  }
}

// ---------------------------------------------------------------------------
// Launcher
// ---------------------------------------------------------------------------
extern "C" void kernel_launch(void* const* d_in, const int* in_sizes, int n_in,
                              void* d_out, int out_size, void* d_ws, size_t ws_size,
                              hipStream_t stream) {
  const float*  x  = (const float*)d_in[0];   // [N, D_IN]
  const float*  W1 = (const float*)d_in[1];   // [D_H, D_IN]
  const float*  b1 = (const float*)d_in[2];   // [D_H]
  const __bf16* W2 = (const __bf16*)d_in[3];  // [D_OUT, D_H]
  const __bf16* b2 = (const __bf16*)d_in[4];  // [D_OUT]

  const int D_H   = in_sizes[2];
  const int D_OUT = in_sizes[4];
  const int D_IN  = in_sizes[1] / D_H;
  const int N     = in_sizes[0] / D_IN;

  char* ws = (char*)d_ws;
  __bf16* xbf  = (__bf16*)ws; ws += (size_t)N * D_IN * sizeof(__bf16);
  __bf16* w1bf = (__bf16*)ws; ws += (size_t)D_H * D_IN * sizeof(__bf16);
  __bf16* hbf  = (__bf16*)ws; ws += (size_t)N * D_H * sizeof(__bf16);
  float*  b2f  = (float*)ws;

  // Stage 0: precision staging (negligible vs GEMM time)
  k_dequant_fp4_w1<<<D_H, 256, 0, stream>>>(W1, w1bf, D_IN);
  k_f32_to_bf16<<<4096, 256, 0, stream>>>(x, xbf, N * D_IN);
  k_bf16_to_f32<<<(D_OUT + 255) / 256, 256, 0, stream>>>(b2, b2f, D_OUT);

  // Stage 1: h = relu(x @ W1_dq^T + b1) -> bf16
  dim3 g1(D_H / TBN, N / TBM);
  k_gemm_bf16_wmma<true><<<g1, 256, 0, stream>>>(xbf, w1bf, b1, hbf, N, D_H, D_IN);

  // Stage 2: out = h @ W2^T + b2 -> bf16
  dim3 g2(D_OUT / TBN, N / TBM);
  k_gemm_bf16_wmma<false><<<g2, 256, 0, stream>>>(hbf, W2, b2f, (__bf16*)d_out,
                                                  N, D_OUT, D_H);
}
